// VolumeFeatureRenderer_80891414053681
// MI455X (gfx1250) — matrix operand order repaired and checked by reading, not compiled, not run on hardware
//
#include <hip/hip_runtime.h>
#include <hip/hip_bf16.h>
#include <math.h>

typedef _Float16 h16;
typedef __attribute__((ext_vector_type(16))) _Float16 v16h;
typedef __attribute__((ext_vector_type(8)))  _Float16 v8h;
typedef __attribute__((ext_vector_type(8)))  float    v8f;

#define WID 256
#define SDIM 256
#define NS 24
#define ROWS 48          // 2 rays * 24 samples per block
#define NMAT 8           // 7 hidden layers + view layer
#define NT 16            // N tiles of 16
#define KF 8             // K fragments of 32
#define HSTRIDE 264      // padded LDS row stride (halves)
#define THREADS 384      // 12 waves

#define GP_OFF 0         // gamma/gbb params: 9*2*256*2 floats = 36864 B
#define WF_OFF 65536     // f16 weight fragments: 8*16*8*32*16 halves = 1 MB

// ---------------------------------------------------------------------------
// Prep 1: per-layer, per-batch FiLM params folded as (gamma, gamma*bias+beta)
// ---------------------------------------------------------------------------
__global__ void prep_gamma(const float* __restrict__ styles,
    const float* __restrict__ g0w, const float* __restrict__ g0b,
    const float* __restrict__ be0w, const float* __restrict__ be0b,
    const float* __restrict__ b0,
    const float* __restrict__ glw, const float* __restrict__ glb,
    const float* __restrict__ blw, const float* __restrict__ blb,
    const float* __restrict__ bl,
    const float* __restrict__ gvw, const float* __restrict__ gvb,
    const float* __restrict__ bvw, const float* __restrict__ bvb,
    const float* __restrict__ bv,
    float* __restrict__ gp)
{
  int t = blockIdx.x * blockDim.x + threadIdx.x;
  if (t >= 9 * 2 * 256) return;
  int L = t / 512, rem = t % 512, b = rem / 256, n = rem % 256;
  const float *gw, *gb, *bw, *bbp; float bias;
  if (L == 0)      { gw = g0w + n*SDIM; gb = g0b; bw = be0w + n*SDIM; bbp = be0b; bias = b0[n]; }
  else if (L <= 7) { int i = L - 1;
                     gw = glw + (i*WID + n)*SDIM; gb = glb + i*WID;
                     bw = blw + (i*WID + n)*SDIM; bbp = blb + i*WID; bias = bl[i*WID + n]; }
  else             { gw = gvw + n*SDIM; gb = gvb; bw = bvw + n*SDIM; bbp = bvb; bias = bv[n]; }
  const float* st = styles + b*SDIM;
  float sg = 0.f, sb = 0.f;
  for (int c = 0; c < SDIM; ++c) { float s = st[c]; sg += s * gw[c]; sb += s * bw[c]; }
  float gamma = 15.f * (sg + gb[n]) + 30.f;
  float beta  = 0.25f * (sb + bbp[n]);
  gp[((L*2 + b)*256 + n)*2 + 0] = gamma;
  gp[((L*2 + b)*256 + n)*2 + 1] = gamma * bias + beta;
}

// ---------------------------------------------------------------------------
// Prep 2: swizzle fp32 weights into f16 WMMA B fragments.
// B[k][n] = W[n][k].  Fragment layout (16x16x32 f16, wave32):
//   lane l: N = l%16 ; halves hh=0..7 hold K = (l<16 ? 0 : 8) + hh, hh=8..15 hold K = 16 + (l<16 ? 0 : 8) + hh-8
// Linear index = (((mat*16+ntile)*8+kf)*32+lane)*16+hh
// ---------------------------------------------------------------------------
__global__ void prep_wfrag(const float* __restrict__ wl,
                           const float* __restrict__ wv,
                           h16* __restrict__ wf)
{
  int t = blockIdx.x * blockDim.x + threadIdx.x;
  if (t >= NMAT*NT*KF*32*16) return;
  int hh    = t & 15;
  int lane  = (t >> 4) & 31;
  int kf    = (t >> 9) & 7;
  int ntile = (t >> 12) & 15;
  int mat   = t >> 16;
  int n = ntile*16 + (lane & 15);
  int k = kf*32 + ((lane < 16) ? 0 : 8) + ((hh < 8) ? hh : hh + 8);
  float w = (mat < 7) ? wl[(mat*WID + n)*WID + k] : wv[n*(WID+3) + k];
  wf[t] = (h16)w;
}

// ---------------------------------------------------------------------------
// Main: 48 points (2 rays) per block. Full MLP via WMMA + in-block composite.
// ---------------------------------------------------------------------------
__global__ __launch_bounds__(THREADS)
void nerf_main(const float* __restrict__ x,
               const float* __restrict__ z_vals,
               const float* __restrict__ rays_d,
               const float* __restrict__ w0,
               const float* __restrict__ wv,
               const float* __restrict__ w_rgb, const float* __restrict__ b_rgb,
               const float* __restrict__ w_sdf, const float* __restrict__ b_sdf,
               const float* __restrict__ sigmoid_beta,
               const float* __restrict__ gp,
               const h16* __restrict__ wf,
               float* __restrict__ out)
{
  __shared__ h16   hbuf[2][ROWS][HSTRIDE];
  __shared__ float views_s[ROWS][3];
  __shared__ float sdf_s[ROWS];
  __shared__ float rgb_s[ROWS][3];
  __shared__ float red[ROWS][8];
  __shared__ float wgt[ROWS];

  const int tid = threadIdx.x;
  const int bb  = blockIdx.x;
  const int m0  = bb * ROWS;                 // first global point index
  const int batch = m0 / (64*64*NS);         // 98304 pts per batch; 48 | 98304

  // views for this block
  if (tid < ROWS*3) { int r = tid/3, j = tid%3; views_s[r][j] = x[(m0+r)*6 + 3 + j]; }

  // ---- layer 0: 3 -> 256, plain VALU ----
  {
    const float* gpl = gp + (0*2 + batch)*256*2;
    for (int idx = tid; idx < ROWS*WID; idx += THREADS) {
      int r = idx >> 8, n = idx & 255;
      const float* px = x + (m0 + r)*6;
      float acc = px[0]*w0[n*3+0] + px[1]*w0[n*3+1] + px[2]*w0[n*3+2];
      hbuf[0][r][n] = (h16)__sinf(gpl[n*2]*acc + gpl[n*2+1]);
    }
  }
  __syncthreads();

  const int wave = tid >> 5, lane = tid & 31;
  const int rowtile = wave >> 2;             // 0..2 -> rows [rowtile*16, +16)
  const int nset    = wave & 3;              // 0..3 -> 4 N-tiles each
  const int row0    = rowtile * 16;
  const int arow    = row0 + (lane & 15);
  const int k0base  = (lane < 16) ? 0 : 8;   // A layout: lo lanes K 0-7/16-23
  const int rofs    = (lane < 16) ? 0 : 8;   // C/D layout: hi lanes M += 8

  int cur = 0;
  v16h afrag[KF];

  for (int mat = 0; mat < NMAT; ++mat) {
    // Load A fragments (16x32 f16 each) from LDS, ISA register layout.
    #pragma unroll
    for (int kf = 0; kf < KF; ++kf) {
      const h16* rowp = &hbuf[cur][arow][kf*32 + k0base];
      v8h lo = *(const v8h*)(rowp);          // K k0..k0+7
      v8h hi = *(const v8h*)(rowp + 16);     // K k0+16..k0+23
      v16h a;
      *((v8h*)&a)       = lo;
      *(((v8h*)&a) + 1) = hi;
      afrag[kf] = a;
    }
    const int L = mat + 1;                   // FiLM layer index 1..8
    const float* gpl = gp + (L*2 + batch)*256*2;

    #pragma unroll
    for (int t4 = 0; t4 < 4; ++t4) {
      const int ntile = nset*4 + t4;
      const int n = ntile*16 + (lane & 15);
      v8f acc = {0.f,0.f,0.f,0.f,0.f,0.f,0.f,0.f};
      const v16h* bp = (const v16h*)wf + ((size_t)(mat*NT + ntile)*KF)*32 + lane;
      #pragma unroll
      for (int kf = 0; kf < KF; ++kf) {
        v16h b = bp[(size_t)kf * 32];
        acc = __builtin_amdgcn_wmma_f32_16x16x32_f16(
                false, afrag[kf], false, b, (short)0, acc, false, false);
      }
      if (mat == 7) {                        // view layer: add 3 extra columns
        float c0 = wv[n*(WID+3) + 256];
        float c1 = wv[n*(WID+3) + 257];
        float c2 = wv[n*(WID+3) + 258];
        #pragma unroll
        for (int i = 0; i < 8; ++i) {
          int r = row0 + i + rofs;
          acc[i] += views_s[r][0]*c0 + views_s[r][1]*c1 + views_s[r][2]*c2;
        }
      }
      float gamma = gpl[n*2], gbb = gpl[n*2+1];
      #pragma unroll
      for (int i = 0; i < 8; ++i) {
        int r = row0 + i + rofs;
        hbuf[cur ^ 1][r][n] = (h16)__sinf(gamma*acc[i] + gbb);
      }
    }
    __syncthreads();
    cur ^= 1;
  }
  // feat is in hbuf[cur]; pre-view h is in hbuf[cur^1]

  // ---- sdf head: h . w_sdf ----
  {
    int r = tid >> 3, c8 = tid & 7;          // 48 rows x 8 partials
    float s = 0.f;
    const h16* hr = &hbuf[cur ^ 1][r][0];
    for (int j = 0; j < 32; ++j) s += (float)hr[c8*32 + j] * w_sdf[c8*32 + j];
    red[r][c8] = s;
  }
  __syncthreads();
  if (tid < ROWS) {
    float s = b_sdf[0];
    for (int j = 0; j < 8; ++j) s += red[tid][j];
    sdf_s[tid] = s;
  }
  // ---- rgb head: feat . w_rgb ----
  if (tid < ROWS*3) {
    int r = tid/3, j = tid%3;
    float s = b_rgb[j];
    const h16* fr = &hbuf[cur][r][0];
    for (int c = 0; c < WID; ++c) s += (float)fr[c] * w_rgb[j*WID + c];
    rgb_s[r][j] = s;
  }
  __syncthreads();

  // ---- per-ray weights (2 rays, serial 24-step composite) ----
  if (tid < 2) {
    int ray = bb*2 + tid;
    float r0 = rays_d[ray*3+0], r1 = rays_d[ray*3+1], r2 = rays_d[ray*3+2];
    float nrm = sqrtf(r0*r0 + r1*r1 + r2*r2);
    float beta = sigmoid_beta[0];
    float vis = 1.f, wsum = 0.f;
    for (int s = 0; s < NS; ++s) {
      float dist = (s < NS-1) ? (z_vals[ray*NS + s + 1] - z_vals[ray*NS + s]) : 1e10f;
      dist *= nrm;
      float sd = sdf_s[tid*NS + s];
      float sigma = 1.f / ((1.f + __expf(sd / beta)) * beta);  // sigmoid(-sd/b)/b
      float alpha = 1.f - __expf(-sigma * dist);
      if (s < NS-1) {
        float w = alpha * vis;
        wgt[tid*NS + s] = w;
        wsum += w;
        vis *= (1.f - alpha + 1e-10f);
      } else {
        wgt[tid*NS + s] = 1.f - wsum;
      }
    }
  }
  __syncthreads();

  // ---- outputs: rgb_map (3) + feature_map (256) per ray ----
  for (int c = tid; c < 2*WID; c += THREADS) {
    int q = c >> 8, ch = c & 255;
    float a = 0.f;
    for (int s = 0; s < NS; ++s) a += wgt[q*NS + s] * (float)hbuf[cur][q*NS + s][ch];
    out[(size_t)(bb*2 + q)*(WID+3) + 3 + ch] = a;
  }
  if (tid < 6) {
    int q = tid/3, j = tid%3;
    float a = 0.f;
    for (int s = 0; s < NS; ++s)
      a += wgt[q*NS + s] * (1.f / (1.f + __expf(-rgb_s[q*NS + s][j])));
    out[(size_t)(bb*2 + q)*(WID+3) + j] = -1.f + 2.f*a;
  }
}

// ---------------------------------------------------------------------------
extern "C" void kernel_launch(void* const* d_in, const int* in_sizes, int n_in,
                              void* d_out, int out_size, void* d_ws, size_t ws_size,
                              hipStream_t stream)
{
  (void)in_sizes; (void)n_in; (void)out_size; (void)ws_size;
  const float* x       = (const float*)d_in[0];
  const float* styles  = (const float*)d_in[1];
  const float* z_vals  = (const float*)d_in[2];
  const float* rays_d  = (const float*)d_in[3];
  const float* w0      = (const float*)d_in[4];
  const float* b0      = (const float*)d_in[5];
  const float* g0w     = (const float*)d_in[6];
  const float* g0b     = (const float*)d_in[7];
  const float* be0w    = (const float*)d_in[8];
  const float* be0b    = (const float*)d_in[9];
  const float* wl      = (const float*)d_in[10];
  const float* bl      = (const float*)d_in[11];
  const float* glw     = (const float*)d_in[12];
  const float* glb     = (const float*)d_in[13];
  const float* blw     = (const float*)d_in[14];
  const float* blb     = (const float*)d_in[15];
  const float* wv      = (const float*)d_in[16];
  const float* bv      = (const float*)d_in[17];
  const float* gvw     = (const float*)d_in[18];
  const float* gvb     = (const float*)d_in[19];
  const float* bvw     = (const float*)d_in[20];
  const float* bvb     = (const float*)d_in[21];
  const float* w_rgb   = (const float*)d_in[22];
  const float* b_rgb   = (const float*)d_in[23];
  const float* w_sdf   = (const float*)d_in[24];
  const float* b_sdf   = (const float*)d_in[25];
  const float* sbeta   = (const float*)d_in[26];

  float* gpp = (float*)((char*)d_ws + GP_OFF);
  h16*   wfp = (h16*)((char*)d_ws + WF_OFF);

  prep_gamma<<<(9*2*256 + 255)/256, 256, 0, stream>>>(
      styles, g0w, g0b, be0w, be0b, b0,
      glw, glb, blw, blb, bl,
      gvw, gvb, bvw, bvb, bv, gpp);

  prep_wfrag<<<(NMAT*NT*KF*32*16)/256, 256, 0, stream>>>(wl, wv, wfp);

  const int nblocks = (2*64*64*NS) / ROWS;   // 4096
  nerf_main<<<nblocks, THREADS, 0, stream>>>(
      x, z_vals, rays_d, w0, wv, w_rgb, b_rgb, w_sdf, b_sdf, sbeta,
      gpp, wfp, (float*)d_out);
}
